// AttentionPooling_44916767981572
// MI455X (gfx1250) — compile-verified, matrix operand order checked by reading, not run on hardware
//
#include <hip/hip_runtime.h>
#include <hip/hip_bf16.h>

typedef __attribute__((ext_vector_type(16))) _Float16 v16h;
typedef __attribute__((ext_vector_type(8)))  float    v8f;

#define HIDDEN 512
#define HMID   256
#define NSEG   256

// gfx1250 has a hardware tanh transcendental (v_tanh_f32). Use it when the
// builtin exists; otherwise fall back to libm expansion.
__device__ __forceinline__ float hw_tanh(float v) {
#if __has_builtin(__builtin_amdgcn_tanhf)
    return __builtin_amdgcn_tanhf(v);
#else
    return tanhf(v);
#endif
}

// ---------------------------------------------------------------------------
// Kernel 0: zero d_out and the segment accumulators (seg_score | seg_w).
// ---------------------------------------------------------------------------
__global__ void ap_zero_kernel(float* __restrict__ out, float* __restrict__ seg,
                               int out_n) {
    int i = blockIdx.x * blockDim.x + threadIdx.x;
    if (i < out_n) out[i] = 0.0f;
    if (i < 2 * NSEG) seg[i] = 0.0f;
}

// ---------------------------------------------------------------------------
// Kernel 1: pack W1 [512,256] f32 row-major -> f16 WMMA-B fragment order.
// Fragment f = kt*16+nt holds the 32x16 (KxN) tile; per lane 16 contiguous
// halfs: n = nt*16 + (lane&15); k0 = kt*32 + 8*(lane>=16);
// halfs 0..7 -> k0..k0+7, halfs 8..15 -> k0+16..k0+23 (mirrors A layout).
// ---------------------------------------------------------------------------
__global__ void ap_pack_w1_kernel(const float* __restrict__ W1,
                                  _Float16* __restrict__ pw) {
    int t    = blockIdx.x * blockDim.x + threadIdx.x;   // 0..8191
    int frag = t >> 5;
    int lane = t & 31;
    int kt = frag >> 4, nt = frag & 15;
    int n  = (nt << 4) + (lane & 15);
    int k0 = kt * 32 + ((lane >> 4) << 3);
    _Float16* dst = pw + (size_t)frag * 512 + lane * 16;
#pragma unroll
    for (int e = 0; e < 16; ++e) {
        int k = k0 + e + ((e >> 3) << 3);               // 0..7 -> +e ; 8..15 -> +8+e
        dst[e] = (_Float16)W1[(size_t)k * HMID + n];
    }
}

// ---------------------------------------------------------------------------
// Kernel 2: WMMA scores. One wave = 16 rows. LDS-staged packed W1 in 4 phases
// of 64KB (4 ktiles x 16 ntiles). acc[nt] f32, hw tanh + W2 dot + shfl reduce.
// ---------------------------------------------------------------------------
__global__ __launch_bounds__(256)
void ap_scores_kernel(const float* __restrict__ x,
                      const long long* __restrict__ batch,
                      const float* __restrict__ b1,
                      const float* __restrict__ W2,
                      const float* __restrict__ b2,
                      const _Float16* __restrict__ pW1,
                      float* __restrict__ scores,
                      float* __restrict__ seg_score,
                      int N) {
    extern __shared__ char smem[];                      // 64 KB dynamic LDS
    _Float16* sW = (_Float16*)smem;

    const int tid  = threadIdx.x;
    const int lane = tid & 31;
    const int wave = tid >> 5;
    const int rowBlock = blockIdx.x * 8 + wave;
    const int rowBase  = rowBlock * 16;
    const bool active  = (rowBase + 16) <= N;           // N == 200000, multiple of 16

    const int col  = lane & 15;                         // A: row index; C: column index
    const int hsel = lane >> 4;

    v8f acc[16];
#pragma unroll
    for (int i = 0; i < 16; ++i)
#pragma unroll
        for (int j = 0; j < 8; ++j) acc[i][j] = 0.0f;

    const float* xrow = x + (size_t)(rowBase + col) * HIDDEN + (hsel << 3);

    for (int phase = 0; phase < 4; ++phase) {
        if (phase) __syncthreads();
        // cooperative fill: 64KB contiguous chunk of packed W1 -> LDS
        const float4* src = (const float4*)(pW1 + (size_t)phase * 32768);
        const float4* nxt = (const float4*)(pW1 + (size_t)((phase < 3) ? phase + 1 : phase) * 32768);
        float4* dst = (float4*)smem;
#pragma unroll
        for (int i = 0; i < 16; ++i) {
            dst[tid + (i << 8)] = src[tid + (i << 8)];
            __builtin_prefetch(&nxt[tid + (i << 8)], 0, 1);
        }
        __syncthreads();

        if (active) {
#pragma unroll
            for (int ktl = 0; ktl < 4; ++ktl) {
                const int kt = phase * 4 + ktl;
                const float* xp = xrow + kt * 32;
                float4 f0 = *(const float4*)(xp);
                float4 f1 = *(const float4*)(xp + 4);
                float4 f2 = *(const float4*)(xp + 16);
                float4 f3 = *(const float4*)(xp + 20);
                v16h a;
                a[0]  = (_Float16)f0.x; a[1]  = (_Float16)f0.y;
                a[2]  = (_Float16)f0.z; a[3]  = (_Float16)f0.w;
                a[4]  = (_Float16)f1.x; a[5]  = (_Float16)f1.y;
                a[6]  = (_Float16)f1.z; a[7]  = (_Float16)f1.w;
                a[8]  = (_Float16)f2.x; a[9]  = (_Float16)f2.y;
                a[10] = (_Float16)f2.z; a[11] = (_Float16)f2.w;
                a[12] = (_Float16)f3.x; a[13] = (_Float16)f3.y;
                a[14] = (_Float16)f3.z; a[15] = (_Float16)f3.w;

                const _Float16* bbase = sW + (size_t)(ktl * 16) * 512 + lane * 16;
#pragma unroll
                for (int nt = 0; nt < 16; ++nt) {
                    v16h b = *(const v16h*)(bbase + nt * 512);
                    acc[nt] = __builtin_amdgcn_wmma_f32_16x16x32_f16(
                        false, a, false, b, (short)0, acc[nt], false, false);
                }
            }
        }
    }

    if (active) {
        // h = tanh(acc + b1); per-lane partial of h . W2 over this lane's column.
        float p[8];
#pragma unroll
        for (int j = 0; j < 8; ++j) p[j] = 0.0f;
#pragma unroll
        for (int nt = 0; nt < 16; ++nt) {
            const int idx = nt * 16 + col;
            const float b1v = b1[idx];
            const float w2v = W2[idx];
#pragma unroll
            for (int j = 0; j < 8; ++j)
                p[j] += hw_tanh(acc[nt][j] + b1v) * w2v;
        }
        // reduce across the 16 lanes of each half (bit4 untouched by masks 1,2,4,8)
#pragma unroll
        for (int j = 0; j < 8; ++j) {
#pragma unroll
            for (int d = 1; d < 16; d <<= 1)
                p[j] += __shfl_xor(p[j], d, 32);
        }
        if (col == 0) {
            const float b2v = b2[0];
            for (int j = 0; j < 8; ++j) {
                int row = rowBase + hsel * 8 + j;       // VGPR j: M=j (lanes 0-15), M=j+8 (16-31)
                float sc = p[j] + b2v;
                scores[row] = sc;
                atomicAdd(&seg_score[(int)batch[row]], sc);
            }
        }
    }
}

// ---------------------------------------------------------------------------
// Kernel 3: att = exp(score - seg_score[b]); seg_w[b] += att  (scatter-add).
// ---------------------------------------------------------------------------
__global__ void ap_weights_kernel(const float* __restrict__ scores,
                                  const long long* __restrict__ batch,
                                  const float* __restrict__ seg_score,
                                  float* __restrict__ seg_w, int N) {
    int i = blockIdx.x * blockDim.x + threadIdx.x;
    if (i < N) {
        int b = (int)batch[i];
        atomicAdd(&seg_w[b], expf(scores[i] - seg_score[b]));
    }
}

// ---------------------------------------------------------------------------
// Kernel 4: out[b,:] += x[i,:] * w[i]. batch sorted -> running accumulator,
// flush on segment change only. 128 rows/block, 2 cols/thread.
// ---------------------------------------------------------------------------
__global__ __launch_bounds__(256)
void ap_pool_kernel(const float* __restrict__ x,
                    const long long* __restrict__ batch,
                    const float* __restrict__ scores,
                    const float* __restrict__ seg_score,
                    const float* __restrict__ seg_w,
                    float* __restrict__ out, int N) {
    __shared__ float wbuf[128];
    const int tid = threadIdx.x;
    const int r0  = blockIdx.x * 128;
    const int r1  = (r0 + 128 < N) ? (r0 + 128) : N;

    if (tid < 128) {
        int r = r0 + tid;
        if (r < N) {
            int b = (int)batch[r];
            wbuf[tid] = expf(scores[r] - seg_score[b]) / (seg_w[b] + 1e-8f);
        }
    }
    __syncthreads();

    const int c0 = tid * 2;
    float ax = 0.0f, ay = 0.0f;
    int cur = -1;
    for (int r = r0; r < r1; ++r) {
        int b = (int)batch[r];
        if (b != cur) {
            if (cur >= 0) {
                atomicAdd(&out[(size_t)cur * HIDDEN + c0],     ax);
                atomicAdd(&out[(size_t)cur * HIDDEN + c0 + 1], ay);
            }
            ax = 0.0f; ay = 0.0f; cur = b;
        }
        const float w = wbuf[r - r0];
        const float2 xv = *(const float2*)(x + (size_t)r * HIDDEN + c0);
        ax += xv.x * w;
        ay += xv.y * w;
    }
    if (cur >= 0) {
        atomicAdd(&out[(size_t)cur * HIDDEN + c0],     ax);
        atomicAdd(&out[(size_t)cur * HIDDEN + c0 + 1], ay);
    }
}

// ---------------------------------------------------------------------------
extern "C" void kernel_launch(void* const* d_in, const int* in_sizes, int n_in,
                              void* d_out, int out_size, void* d_ws, size_t ws_size,
                              hipStream_t stream) {
    const float*     x     = (const float*)d_in[0];
    const long long* batch = (const long long*)d_in[1];
    const float*     W1    = (const float*)d_in[2];
    const float*     b1    = (const float*)d_in[3];
    const float*     W2    = (const float*)d_in[4];
    const float*     b2    = (const float*)d_in[5];
    float*           out   = (float*)d_out;

    const int N = in_sizes[1];                          // 200000

    // workspace layout: scores[N] | seg_score[256] | seg_w[256] | packedW1(f16)
    char* ws = (char*)d_ws;
    float* scores    = (float*)ws;
    float* seg_score = scores + N;
    float* seg_w     = seg_score + NSEG;
    size_t packOff   = (((size_t)N * 4 + 2 * NSEG * 4) + 255) & ~(size_t)255;
    _Float16* pW1    = (_Float16*)(ws + packOff);       // 131072 halfs = 256 KB

    // 0) zero output + segment accumulators
    {
        int total = out_size;                            // 131072 >= 512
        int blocks = (total + 255) / 256;
        ap_zero_kernel<<<blocks, 256, 0, stream>>>(out, seg_score, total);
    }
    // 1) pack W1 -> f16 fragments (8192 threads)
    ap_pack_w1_kernel<<<32, 256, 0, stream>>>(W1, pW1);

    // 2) WMMA scores: 12500 row tiles, 8 waves/block
    {
        int rowBlocks = (N + 15) / 16;
        int blocks = (rowBlocks + 7) / 8;
        ap_scores_kernel<<<blocks, 256, 65536, stream>>>(
            x, batch, b1, W2, b2, pW1, scores, seg_score, N);
    }
    // 3) segment exp-weight sums
    ap_weights_kernel<<<(N + 255) / 256, 256, 0, stream>>>(
        scores, batch, seg_score, seg_w, N);

    // 4) weighted pooled output
    ap_pool_kernel<<<(N + 127) / 128, 256, 0, stream>>>(
        x, batch, scores, seg_score, seg_w, out, N);
}
